// Model_86835648790645
// MI455X (gfx1250) — matrix operand order repaired and checked by reading, not compiled, not run on hardware
//
#include <hip/hip_runtime.h>
#include <hip/hip_bf16.h>
#include <math.h>

typedef __attribute__((ext_vector_type(16))) _Float16 v16h;
typedef __attribute__((ext_vector_type(8)))  _Float16 v8h;
typedef __attribute__((ext_vector_type(8)))  float    v8f;
typedef __attribute__((ext_vector_type(4)))  float    v4f;

union F16Frag { v16h v; v8h p[2]; };

#define EPSV 1e-6f
#define S_LD 48      // leading dim of node scalar buffers
#define V_LD 24      // 8 vectors * 3
#define AGG_S_LD 56  // max a+b
#define AGG_V_LD 192 // max (a+2b)*3

// ---------------------------------------------------------------------------
// Weight packing into WMMA B-fragment order:
//   frag element (tile t, lane l, half h) = W[K][col],
//   K = (h&7) + 8*(l>>4) + 16*(h>>3), col = t*16 + (l&15)
// W1 (16x32, K padded to 32 with zeros) -> 2 tiles  (1024 halves)
// W2 (32xOUT, col padded to 128)        -> 8 tiles  (4096 halves)
// ---------------------------------------------------------------------------
__global__ void pack_weights_kernel(const float* __restrict__ W1,
                                    const float* __restrict__ W2,
                                    _Float16* __restrict__ w1f,
                                    _Float16* __restrict__ w2f, int OUT) {
  int i = blockIdx.x * blockDim.x + threadIdx.x;
  if (i < 2 * 32 * 16) {
    int h = i & 15, l = (i >> 4) & 31, t = i >> 9;
    int K = (h & 7) + 8 * (l >> 4) + 16 * (h >> 3);
    int col = t * 16 + (l & 15);
    w1f[i] = (_Float16)((K < 16) ? W1[K * 32 + col] : 0.0f);
  }
  if (i < 8 * 32 * 16) {
    int h = i & 15, l = (i >> 4) & 31, t = i >> 9;
    int K = (h & 7) + 8 * (l >> 4) + 16 * (h >> 3);
    int col = t * 16 + (l & 15);
    w2f[i] = (_Float16)((col < OUT) ? W2[K * OUT + col] : 0.0f);
  }
}

// ---------------------------------------------------------------------------
// Node init: s = concat(feat_s, emb[residue_type]), v = 0
// ---------------------------------------------------------------------------
__global__ void init_nodes_kernel(const float* __restrict__ feat_s,
                                  const int* __restrict__ residue_type,
                                  const float* __restrict__ emb,
                                  float* __restrict__ sA, float* __restrict__ vA,
                                  int N) {
  int n = blockIdx.x * blockDim.x + threadIdx.x;
  if (n >= N) return;
  int rt = residue_type[n];
  for (int j = 0; j < 32; ++j) sA[(size_t)n * S_LD + j] = feat_s[(size_t)n * 32 + j];
  for (int j = 0; j < 16; ++j) sA[(size_t)n * S_LD + 32 + j] = emb[rt * 16 + j];
  for (int j = 0; j < V_LD; ++j) vA[(size_t)n * V_LD + j] = 0.0f;
}

// ---------------------------------------------------------------------------
// Zero the aggregation buffers (required before atomic scatter)
// ---------------------------------------------------------------------------
__global__ void zero_agg_kernel(float* __restrict__ agg_s, float* __restrict__ agg_v,
                                int N) {
  size_t stride = (size_t)gridDim.x * blockDim.x;
  size_t gid = (size_t)blockIdx.x * blockDim.x + threadIdx.x;
  size_t ns = (size_t)N * AGG_S_LD, nv = (size_t)N * AGG_V_LD;
  for (size_t i = gid; i < ns; i += stride) agg_s[i] = 0.0f;
  for (size_t i = gid; i < nv; i += stride) agg_v[i] = 0.0f;
}

// ---------------------------------------------------------------------------
// Edge kernel: one 16-edge tile per wave, 4 waves / 128 threads per block.
//   phi(d) [16x32 padded f16] @ W1 [32x32] -> relu -> hid [16x32 f16]
//   hid @ W2 [32x128 padded] -> w [16xOUT]
// Both GEMMs via v_wmma_f32_16x16x32_f16; fragments move through LDS with
// b128 vector accesses. Messages scatter into agg buffers with f32 atomics.
// ---------------------------------------------------------------------------
template <int A, int B>
__global__ void __launch_bounds__(128)
edge_kernel(const float* __restrict__ pos,
            const int* __restrict__ esrc, const int* __restrict__ edst,
            const float* __restrict__ sA, const float* __restrict__ vA,
            const _Float16* __restrict__ w1f_g, const _Float16* __restrict__ w2f_g,
            const float* __restrict__ b1, const float* __restrict__ b2,
            float* __restrict__ agg_s, float* __restrict__ agg_v, int E_) {
  constexpr int OUT = 2 * A + 3 * B;
  constexpr int OUT_TILES = (OUT + 15) / 16;

  __shared__ __attribute__((aligned(32))) _Float16 sW1f[2 * 32 * 16];
  __shared__ __attribute__((aligned(32))) _Float16 sW2f[8 * 32 * 16];
  __shared__ __attribute__((aligned(32))) _Float16 sPhi[4][16 * 32];
  __shared__ __attribute__((aligned(32))) _Float16 sHid[4][16 * 32];
  __shared__ __attribute__((aligned(32))) float    sWoutT[4][128 * 16]; // [col][edge]
  __shared__ float sU[4][16][4];
  __shared__ int   sSrc[4][16];
  __shared__ int   sDst[4][16];

  const int t = threadIdx.x;
  const int wave = t >> 5;
  const int lane = t & 31;
  const int em = lane & 15;  // edge-in-tile (row M)
  const int gg = lane >> 4;  // lane-group (layout half / work split)

  const int eb = blockIdx.x * 64 + wave * 16;

  // Prefetch edge indices 256 edges (4 blocks) ahead (global_prefetch_b8 path)
  if (t == 0 && eb + 256 < E_) {
    __builtin_prefetch(esrc + eb + 256, 0, 1);
    __builtin_prefetch(edst + eb + 256, 0, 1);
  }

  // Stage packed weights into LDS with 128-bit copies (shared by all waves)
  {
    const uint4* g1 = (const uint4*)w1f_g;
    uint4* s1 = (uint4*)sW1f;
    if (t < 128) s1[t] = g1[t];
    const uint4* g2 = (const uint4*)w2f_g;
    uint4* s2 = (uint4*)sW2f;
    for (int i = t; i < 512; i += 128) s2[i] = g2[i];
  }

  // --- stage edge geometry + gaussian basis (lanes 0..15, one edge each) ---
  if (lane < 16) {
    int e = eb + lane;
    bool valid = (e < E_);
    int s_ = valid ? esrc[e] : 0;
    int d_ = valid ? edst[e] : 0;
    float rx = pos[(size_t)d_ * 3 + 0] - pos[(size_t)s_ * 3 + 0];
    float ry = pos[(size_t)d_ * 3 + 1] - pos[(size_t)s_ * 3 + 1];
    float rz = pos[(size_t)d_ * 3 + 2] - pos[(size_t)s_ * 3 + 2];
    float dd = sqrtf(rx * rx + ry * ry + rz * rz + EPSV);
    float inv = 1.0f / dd;
    sU[wave][lane][0] = rx * inv;
    sU[wave][lane][1] = ry * inv;
    sU[wave][lane][2] = rz * inv;
    sSrc[wave][lane] = s_;
    sDst[wave][lane] = valid ? d_ : -1;
    // phi row (K=0..15 gaussians, K=16..31 zero padding), packed b128 stores
    v8h ph0, ph1;
#pragma unroll
    for (int k = 0; k < 8; ++k) {
      float mu0 = 3.5f * (float)k / 15.0f;
      float mu1 = 3.5f * (float)(k + 8) / 15.0f;
      float x0 = dd - mu0, x1 = dd - mu1;
      ph0[k] = (_Float16)(valid ? __expf(-8.0f * x0 * x0) : 0.0f);
      ph1[k] = (_Float16)(valid ? __expf(-8.0f * x1 * x1) : 0.0f);
    }
    v8h zz = {};
    *(v8h*)&sPhi[wave][lane * 32 + 0] = ph0;
    *(v8h*)&sPhi[wave][lane * 32 + 8] = ph1;
    *(v8h*)&sPhi[wave][lane * 32 + 16] = zz;
    *(v8h*)&sPhi[wave][lane * 32 + 24] = zz;
  }
  __syncthreads();

  // --- GEMM 1: hid = relu(phi @ W1 + b1) ---
  // A fragment: two contiguous 8-half runs per CDNA5 16-bit A layout
  F16Frag a1;
  a1.p[0] = *(const v8h*)&sPhi[wave][em * 32 + 8 * gg];
  a1.p[1] = *(const v8h*)&sPhi[wave][em * 32 + 16 + 8 * gg];
#pragma unroll
  for (int tile = 0; tile < 2; ++tile) {
    v16h bfrag = *(const v16h*)&sW1f[(tile * 32 + lane) * 16];
    v8f c = {};
    c = __builtin_amdgcn_wmma_f32_16x16x32_f16(false, a1.v, false, bfrag,
                                               (short)0, c, false, false);
    int col = tile * 16 + em;
    float bias = b1[col];
#pragma unroll
    for (int j = 0; j < 8; ++j) {
      float v = c[j] + bias;
      v = v > 0.0f ? v : 0.0f;
      int M = j + 8 * gg;  // C layout: VGPR j holds rows M=j / M=j+8
      sHid[wave][M * 32 + col] = (_Float16)v;
    }
  }
  __syncthreads();

  // --- GEMM 2: w = hid @ W2 + b2 (K = 32 exactly matches 16x16x32) ---
  F16Frag a2;
  a2.p[0] = *(const v8h*)&sHid[wave][em * 32 + 8 * gg];
  a2.p[1] = *(const v8h*)&sHid[wave][em * 32 + 16 + 8 * gg];
#pragma unroll
  for (int tile = 0; tile < OUT_TILES; ++tile) {
    v16h bfrag = *(const v16h*)&sW2f[(tile * 32 + lane) * 16];
    v8f c = {};
    c = __builtin_amdgcn_wmma_f32_16x16x32_f16(false, a2.v, false, bfrag,
                                               (short)0, c, false, false);
    int col = tile * 16 + em;
    float bias = (col < OUT) ? b2[col] : 0.0f;
    // transposed store: 8 consecutive floats -> 2x ds_store_b128
    v4f lo = {c[0] + bias, c[1] + bias, c[2] + bias, c[3] + bias};
    v4f hi = {c[4] + bias, c[5] + bias, c[6] + bias, c[7] + bias};
    *(v4f*)&sWoutT[wave][col * 16 + 8 * gg + 0] = lo;
    *(v4f*)&sWoutT[wave][col * 16 + 8 * gg + 4] = hi;
  }
  __syncthreads();

#define WVAL(j) sWoutT[wave][(j) * 16 + em]

  // --- message formation + scatter (2 lanes cooperate per edge) ---
  int dst = sDst[wave][em];
  if (dst >= 0) {
    int src = sSrc[wave][em];
    float ux = sU[wave][em][0], uy = sU[wave][em][1], uz = sU[wave][em][2];
    float* as_ = agg_s + (size_t)dst * AGG_S_LD;
    float* av_ = agg_v + (size_t)dst * AGG_V_LD;
    const float* sj = sA + (size_t)src * S_LD;
    const float* vj = vA + (size_t)src * V_LD;

    for (int j = gg; j < A; j += 2) {
      float sjv = sj[j];
      atomicAdd(&as_[j], WVAL(j) * sjv);           // w_ss * s_j
      float coef = WVAL(A + j) * sjv;              // w_sv * s_j * u
      atomicAdd(&av_[j * 3 + 0], coef * ux);
      atomicAdd(&av_[j * 3 + 1], coef * uy);
      atomicAdd(&av_[j * 3 + 2], coef * uz);
    }
    for (int j = gg; j < B; j += 2) {
      float vx = vj[j * 3 + 0], vy = vj[j * 3 + 1], vz = vj[j * 3 + 2];
      float dot = vx * ux + vy * uy + vz * uz;
      atomicAdd(&as_[A + j], WVAL(2 * A + j) * dot);  // w_vs * (v_j . u)
      float w0 = WVAL(2 * A + B + j);                 // w_vv0 * v_j
      atomicAdd(&av_[(A + j) * 3 + 0], w0 * vx);
      atomicAdd(&av_[(A + j) * 3 + 1], w0 * vy);
      atomicAdd(&av_[(A + j) * 3 + 2], w0 * vz);
      float w1 = WVAL(2 * A + 2 * B + j);             // w_vv1 * cross(v_j, u)
      float cx = vy * uz - vz * uy;
      float cy = vz * ux - vx * uz;
      float cz = vx * uy - vy * ux;
      atomicAdd(&av_[(A + B + j) * 3 + 0], w1 * cx);
      atomicAdd(&av_[(A + B + j) * 3 + 1], w1 * cy);
      atomicAdd(&av_[(A + B + j) * 3 + 2], w1 * cz);
    }
  }
#undef WVAL
}

// ---------------------------------------------------------------------------
// Node update: s_out = 0.25*agg_s @ Ws + bs ; v_out = 0.25*agg_v contracted Wv
// MODE 0: layer0  -> relu, save raw (sB,vB), save eq_norm (sA,vA)
// MODE 1: layer1/2-> relu, +residual(sB,vB), save raw, save eq_norm
// MODE 2: layer3  -> raw only into sB,vB
// ---------------------------------------------------------------------------
template <int A, int B, int C, int D, int MODE>
__global__ void node_kernel(const float* __restrict__ agg_s,
                            const float* __restrict__ agg_v,
                            const float* __restrict__ Ws, const float* __restrict__ bs,
                            const float* __restrict__ Wv,
                            float* __restrict__ sA, float* __restrict__ vA,
                            float* __restrict__ sB, float* __restrict__ vB, int N) {
  int n = blockIdx.x * blockDim.x + threadIdx.x;
  if (n >= N) return;
  const float* as_ = agg_s + (size_t)n * AGG_S_LD;
  const float* av_ = agg_v + (size_t)n * AGG_V_LD;

  float so[C];
#pragma unroll
  for (int j = 0; j < C; ++j) so[j] = bs[j];
  for (int k = 0; k < A + B; ++k) {
    float x = 0.25f * as_[k];
#pragma unroll
    for (int j = 0; j < C; ++j) so[j] += x * Ws[k * C + j];
  }

  float vo[D][3];
#pragma unroll
  for (int j = 0; j < D; ++j)
    for (int c = 0; c < 3; ++c) vo[j][c] = 0.0f;
  for (int k = 0; k < A + 2 * B; ++k) {
    float a0 = 0.25f * av_[k * 3 + 0];
    float a1 = 0.25f * av_[k * 3 + 1];
    float a2 = 0.25f * av_[k * 3 + 2];
#pragma unroll
    for (int j = 0; j < D; ++j) {
      float wk = Wv[k * D + j];
      vo[j][0] += wk * a0;
      vo[j][1] += wk * a1;
      vo[j][2] += wk * a2;
    }
  }

  if (MODE == 2) {
#pragma unroll
    for (int j = 0; j < C; ++j) sB[(size_t)n * S_LD + j] = so[j];
#pragma unroll
    for (int j = 0; j < D; ++j)
      for (int c = 0; c < 3; ++c) vB[(size_t)n * V_LD + j * 3 + c] = vo[j][c];
    return;
  }

#pragma unroll
  for (int j = 0; j < C; ++j) so[j] = fmaxf(so[j], 0.0f);
  if (MODE == 1) {
#pragma unroll
    for (int j = 0; j < C; ++j) so[j] += sB[(size_t)n * S_LD + j];
#pragma unroll
    for (int j = 0; j < D; ++j)
      for (int c = 0; c < 3; ++c) vo[j][c] += vB[(size_t)n * V_LD + j * 3 + c];
  }
  // save raw (residual source for next layer)
#pragma unroll
  for (int j = 0; j < C; ++j) sB[(size_t)n * S_LD + j] = so[j];
#pragma unroll
  for (int j = 0; j < D; ++j)
    for (int c = 0; c < 3; ++c) vB[(size_t)n * V_LD + j * 3 + c] = vo[j][c];

  // eq_norm -> conv input buffers
  float mean = 0.0f;
#pragma unroll
  for (int j = 0; j < C; ++j) mean += so[j];
  mean *= (1.0f / C);
  float var = 0.0f;
#pragma unroll
  for (int j = 0; j < C; ++j) {
    float d = so[j] - mean;
    var += d * d;
  }
  var *= (1.0f / C);
  float rs = rsqrtf(var + EPSV);
#pragma unroll
  for (int j = 0; j < C; ++j) sA[(size_t)n * S_LD + j] = (so[j] - mean) * rs;

  float ss = 0.0f;
#pragma unroll
  for (int j = 0; j < D; ++j)
    for (int c = 0; c < 3; ++c) ss += vo[j][c] * vo[j][c];
  float vn = sqrtf(ss * (1.0f / D) + EPSV);
  float inv = 1.0f / vn;
#pragma unroll
  for (int j = 0; j < D; ++j)
    for (int c = 0; c < 3; ++c) vA[(size_t)n * V_LD + j * 3 + c] = vo[j][c] * inv;
}

// ---------------------------------------------------------------------------
// Structure builder: frames, torsion rotations, transform chain, atoms
// ---------------------------------------------------------------------------
__global__ void structure_kernel(const float* __restrict__ pos,
                                 const int* __restrict__ residue_type,
                                 const float* __restrict__ sB,
                                 const float* __restrict__ vB,
                                 const float* __restrict__ Ttab,
                                 const int* __restrict__ tdep,
                                 const float* __restrict__ rig,
                                 const int* __restrict__ rdep,
                                 float* __restrict__ out, int N) {
  int n = blockIdx.x * blockDim.x + threadIdx.x;
  if (n >= N) return;
  int rt = residue_type[n];

  float cs[7], sn[7];
#pragma unroll
  for (int i = 0; i < 7; ++i) {
    float x = sB[(size_t)n * S_LD + 2 * i];
    float y = sB[(size_t)n * S_LD + 2 * i + 1];
    float r = rsqrtf(x * x + y * y + EPSV);
    cs[i] = x * r;
    sn[i] = y * r;
  }
  float v0[3], v1[3], v2[3];
#pragma unroll
  for (int c = 0; c < 3; ++c) {
    v0[c] = vB[(size_t)n * V_LD + c];
    v1[c] = vB[(size_t)n * V_LD + 3 + c];
    v2[c] = vB[(size_t)n * V_LD + 6 + c];
  }
  float e1[3], e2[3], e3[3];
  {
    float r = rsqrtf(v0[0] * v0[0] + v0[1] * v0[1] + v0[2] * v0[2] + EPSV);
    for (int c = 0; c < 3; ++c) e1[c] = v0[c] * r;
  }
  float dot = e1[0] * v1[0] + e1[1] * v1[1] + e1[2] * v1[2];
  float u2[3];
  for (int c = 0; c < 3; ++c) u2[c] = v1[c] - dot * e1[c];
  {
    float r = rsqrtf(u2[0] * u2[0] + u2[1] * u2[1] + u2[2] * u2[2] + EPSV);
    for (int c = 0; c < 3; ++c) e2[c] = u2[c] * r;
  }
  e3[0] = e1[1] * e2[2] - e1[2] * e2[1];
  e3[1] = e1[2] * e2[0] - e1[0] * e2[2];
  e3[2] = e1[0] * e2[1] - e1[1] * e2[0];

  float opr[8][4][3];
#pragma unroll
  for (int c = 0; c < 3; ++c) {
    opr[0][0][c] = e1[c];
    opr[0][1][c] = e2[c];
    opr[0][2][c] = e3[c];
    opr[0][3][c] = v2[c] + pos[(size_t)n * 3 + c];
  }
#pragma unroll
  for (int i = 1; i < 8; ++i) {
    opr[i][0][0] = 1.0f; opr[i][0][1] = 0.0f; opr[i][0][2] = 0.0f;
    opr[i][1][0] = 0.0f; opr[i][1][1] = cs[i - 1]; opr[i][1][2] = -sn[i - 1];
    opr[i][2][0] = 0.0f; opr[i][2][1] = sn[i - 1]; opr[i][2][2] = cs[i - 1];
    opr[i][3][0] = 0.0f; opr[i][3][1] = 0.0f; opr[i][3][2] = 0.0f;
  }
  // opr = combine(T[rt], opr)
#pragma unroll
  for (int i = 0; i < 8; ++i) {
    const float* X = Ttab + ((size_t)rt * 8 + i) * 12;
    float R[3][3], tr[3];
    for (int r = 0; r < 3; ++r) {
      for (int c = 0; c < 3; ++c)
        R[r][c] = X[r * 3 + 0] * opr[i][0][c] + X[r * 3 + 1] * opr[i][1][c] +
                  X[r * 3 + 2] * opr[i][2][c];
      tr[r] = X[r * 3 + 0] * opr[i][3][0] + X[r * 3 + 1] * opr[i][3][1] +
              X[r * 3 + 2] * opr[i][3][2] + X[9 + r];
    }
    for (int r = 0; r < 3; ++r) {
      for (int c = 0; c < 3; ++c) opr[i][r][c] = R[r][c];
      opr[i][3][r] = tr[r];
    }
  }
  // dependency chain (tdep[:,i] < i)
  for (int i = 1; i < 8; ++i) {
    int p = tdep[rt * 8 + i];
    float R[3][3], tr[3];
    for (int r = 0; r < 3; ++r) {
      for (int c = 0; c < 3; ++c)
        R[r][c] = opr[p][r][0] * opr[i][0][c] + opr[p][r][1] * opr[i][1][c] +
                  opr[p][r][2] * opr[i][2][c];
      tr[r] = opr[p][r][0] * opr[i][3][0] + opr[p][r][1] * opr[i][3][1] +
              opr[p][r][2] * opr[i][3][2] + opr[p][3][r];
    }
    for (int r = 0; r < 3; ++r) {
      for (int c = 0; c < 3; ++c) opr[i][r][c] = R[r][c];
      opr[i][3][r] = tr[r];
    }
  }
  // atoms
  for (int a = 0; a < 14; ++a) {
    int rd = rdep[rt * 14 + a];
    const float* g = rig + ((size_t)rt * 14 + a) * 3;
    for (int r = 0; r < 3; ++r) {
      out[(size_t)n * 42 + a * 3 + r] =
          opr[rd][r][0] * g[0] + opr[rd][r][1] * g[1] + opr[rd][r][2] * g[2] +
          opr[rd][3][r];
    }
  }
}

// ---------------------------------------------------------------------------
extern "C" void kernel_launch(void* const* d_in, const int* in_sizes, int n_in,
                              void* d_out, int out_size, void* d_ws, size_t ws_size,
                              hipStream_t stream) {
  const float* pos = (const float*)d_in[0];
  const float* feat_s = (const float*)d_in[1];
  const int* residue_type = (const int*)d_in[2];
  const int* edge_src = (const int*)d_in[3];
  const int* edge_dst = (const int*)d_in[4];
  const float* emb = (const float*)d_in[5];
  const int N = in_sizes[0] / 3;
  const int E = in_sizes[3];

  const float *W1[4], *b1[4], *W2[4], *b2[4], *Ws[4], *bs[4], *Wv[4];
  for (int l = 0; l < 4; ++l) {
    int base = 6 + l * 7;
    W1[l] = (const float*)d_in[base + 0];
    b1[l] = (const float*)d_in[base + 1];
    W2[l] = (const float*)d_in[base + 2];
    b2[l] = (const float*)d_in[base + 3];
    Ws[l] = (const float*)d_in[base + 4];
    bs[l] = (const float*)d_in[base + 5];
    Wv[l] = (const float*)d_in[base + 6];
  }
  const float* Ttab = (const float*)d_in[34];
  const int* tdep = (const int*)d_in[35];
  const float* rig = (const float*)d_in[36];
  const int* rdep = (const int*)d_in[37];

  // Workspace carve-up (floats first, then f16 packed-weight fragments)
  char* ws = (char*)d_ws;
  float* sA = (float*)ws;  ws += sizeof(float) * (size_t)N * S_LD;
  float* vA = (float*)ws;  ws += sizeof(float) * (size_t)N * V_LD;
  float* sB = (float*)ws;  ws += sizeof(float) * (size_t)N * S_LD;
  float* vB = (float*)ws;  ws += sizeof(float) * (size_t)N * V_LD;
  float* agg_s = (float*)ws; ws += sizeof(float) * (size_t)N * AGG_S_LD;
  float* agg_v = (float*)ws; ws += sizeof(float) * (size_t)N * AGG_V_LD;
  _Float16* w1f[4];
  _Float16* w2f[4];
  for (int l = 0; l < 4; ++l) {
    w1f[l] = (_Float16*)ws; ws += sizeof(_Float16) * 2 * 32 * 16;
    w2f[l] = (_Float16*)ws; ws += sizeof(_Float16) * 8 * 32 * 16;
  }

  const int outdims[4] = {2 * 48 + 3 * 8, 2 * 32 + 3 * 8, 2 * 32 + 3 * 8, 2 * 32 + 3 * 8};
  for (int l = 0; l < 4; ++l)
    pack_weights_kernel<<<16, 256, 0, stream>>>(W1[l], W2[l], w1f[l], w2f[l], outdims[l]);

  const int nodeBlocks = (N + 127) / 128;
  const int edgeBlocks = (E + 63) / 64;

  init_nodes_kernel<<<nodeBlocks, 128, 0, stream>>>(feat_s, residue_type, emb, sA, vA, N);

  // layer 0 (a=48, b=8)
  zero_agg_kernel<<<1024, 256, 0, stream>>>(agg_s, agg_v, N);
  edge_kernel<48, 8><<<edgeBlocks, 128, 0, stream>>>(
      pos, edge_src, edge_dst, sA, vA, w1f[0], w2f[0], b1[0], b2[0], agg_s, agg_v, E);
  node_kernel<48, 8, 32, 8, 0><<<nodeBlocks, 128, 0, stream>>>(
      agg_s, agg_v, Ws[0], bs[0], Wv[0], sA, vA, sB, vB, N);

  // layers 1, 2 (a=32, b=8, residual)
  for (int l = 1; l <= 2; ++l) {
    zero_agg_kernel<<<1024, 256, 0, stream>>>(agg_s, agg_v, N);
    edge_kernel<32, 8><<<edgeBlocks, 128, 0, stream>>>(
        pos, edge_src, edge_dst, sA, vA, w1f[l], w2f[l], b1[l], b2[l], agg_s, agg_v, E);
    node_kernel<32, 8, 32, 8, 1><<<nodeBlocks, 128, 0, stream>>>(
        agg_s, agg_v, Ws[l], bs[l], Wv[l], sA, vA, sB, vB, N);
  }

  // layer 3 (a=32, b=8, c=14, d=3)
  zero_agg_kernel<<<1024, 256, 0, stream>>>(agg_s, agg_v, N);
  edge_kernel<32, 8><<<edgeBlocks, 128, 0, stream>>>(
      pos, edge_src, edge_dst, sA, vA, w1f[3], w2f[3], b1[3], b2[3], agg_s, agg_v, E);
  node_kernel<32, 8, 14, 3, 2><<<nodeBlocks, 128, 0, stream>>>(
      agg_s, agg_v, Ws[3], bs[3], Wv[3], sA, vA, sB, vB, N);

  structure_kernel<<<nodeBlocks, 128, 0, stream>>>(
      pos, residue_type, sB, vB, Ttab, tdep, rig, rdep, (float*)d_out, N);
}
